// RsLstmModel_16587163697222
// MI455X (gfx1250) — matrix operand order repaired and checked by reading, not compiled, not run on hardware
//
#include <hip/hip_runtime.h>
#include <hip/hip_bf16.h>

typedef __bf16 bf16_t;
typedef __attribute__((ext_vector_type(16))) __bf16 v16bf;
typedef __attribute__((ext_vector_type(4)))  __bf16 v4bf;
typedef __attribute__((ext_vector_type(8)))  float  v8f;

#define HDIM 128   // hidden size
#define GDIM 512   // 4*H gate width
#define TDIM 512   // sequence length
#define BDIM 256   // batch
#define CDIM 12    // classes
#define DIN0 64    // layer-0 input width
#define DIN1 256   // layer-1 input width (2H)
#define BT   16    // batch tile per workgroup (one WMMA M tile)

// ---------------------------------------------------------------------------
// A-fragment packing: map (m, k) of a 16xK bf16 A-matrix to the CDNA5 WMMA
// per-lane layout (ISA 7.12.2, 16-bit A 16x32):
//   lanes 0-15 : K = 0..7 (e=0..7), K = 16..23 (e=8..15)
//   lanes16-31 : K = 8..15,          K = 24..31
// offset (bf16 units) = (k/32)*512 + lane*16 + e
// Property used for vectorization: any 8 consecutive k within an 8-aligned
// group map to 8 contiguous bf16 in LDS (16 bytes).
// ---------------------------------------------------------------------------
__device__ __forceinline__ int packA_off(int m, int k) {
    int blk  = k >> 5;
    int kp   = k & 31;
    int half = (kp >> 3) & 1;               // 0 for [0,8)&[16,24), 1 otherwise
    int e    = (kp & 7) | ((kp & 16) >> 1); // +8 when kp >= 16
    return blk * 512 + ((half << 4) + m) * 16 + e;
}

// ---------------------------------------------------------------------------
// Pack a (N4 x Kd) f32 row-major weight W (as stored: (4H, Din)) into bf16
// B-fragments of B = W^T (shape Kd x N4) for v_wmma_f32_16x16x32_bf16.
// Dense 32x16 B layout: lanes 0-15 hold K=0..7,16..23 (e=0..15), lanes 16-31 hold
// K=8..15,24..31 (same order as A); n = ntile*16 + lane%16.
// Block order: (nt * KT + kt) so the inner kt loop streams contiguous 1KB.
// ---------------------------------------------------------------------------
__global__ void pack_b_kernel(const float* __restrict__ W,
                              bf16_t* __restrict__ dst, int N4, int Kd) {
    int total = N4 * Kd;
    int KT    = Kd >> 5;
    for (int li = blockIdx.x * blockDim.x + threadIdx.x; li < total;
         li += gridDim.x * blockDim.x) {
        int blk  = li >> 9;
        int r    = li & 511;
        int lane = r >> 4;
        int e    = r & 15;
        int nt   = blk / KT;
        int kt   = blk - nt * KT;
        int n    = nt * 16 + (lane & 15);
        int k    = kt * 32 + 8 * (lane >> 4) + e + (e & 8);
        dst[li]  = (bf16_t)W[n * Kd + k];
    }
}

// ---------------------------------------------------------------------------
// Fused BiLSTM layer. One workgroup = one (16-row batch tile, direction).
// 8 waves; wave w owns gate columns {w*16 + 128*j : j=0..3} so the i/f/g/o
// values for h-columns [w*16, w*16+16) stay wave-local. Whh^T fragments are
// register-resident across all T steps (persistent RNN).
// ---------------------------------------------------------------------------
template <int DIN, bool LAYER1>
__global__ __launch_bounds__(256, 1)
void lstm_kernel(const float* __restrict__ X32,   // layer0 input (B,T,DIN0) f32
                 const bf16_t* __restrict__ Xbf,  // layer1 input (B,T,2H) bf16
                 const bf16_t* __restrict__ WihP, // packed, per-dir stride G*DIN
                 const bf16_t* __restrict__ WhhP, // packed, per-dir stride G*H
                 const float* __restrict__ bias,  // (2, 4H)
                 const int* __restrict__ lens,    // (B,)
                 bf16_t* __restrict__ hsOut,      // layer0: (B,T,2H) bf16
                 float* __restrict__ hTOut)       // layer1: (2,B,H) f32
{
    const int dir = blockIdx.y;
    const int b0  = blockIdx.x * BT;
    const int tid = threadIdx.x;
    const int w   = tid >> 5;   // wave id 0..7
    const int L   = tid & 31;   // lane id
    constexpr int KTI = DIN / 32;

    __shared__ __attribute__((aligned(32))) bf16_t sX[KTI * 512]; // x A-frags
    __shared__ __attribute__((aligned(32))) bf16_t sH[4 * 512];   // h A-frags
    __shared__ int sLen[BT];

    const bf16_t* Wih  = WihP + (size_t)dir * GDIM * DIN;
    const bf16_t* Whh  = WhhP + (size_t)dir * GDIM * HDIM;
    const float*  bvec = bias + dir * GDIM;

    if (tid < BT) sLen[tid] = lens[b0 + tid];
    for (int i = tid; i < 4 * 512; i += 256) sH[i] = (bf16_t)0.0f;

    // Register-resident recurrent weight fragments: whh[j][kt]
    v16bf whh[4][4];
#pragma unroll
    for (int j = 0; j < 4; ++j) {
        int nt = w + 8 * j;
#pragma unroll
        for (int kt = 0; kt < 4; ++kt)
            whh[j][kt] = *(const v16bf*)(Whh + (size_t)(nt * 4 + kt) * 512 + L * 16);
    }

    const int n    = L & 15;
    const int hcol = w * 16 + n; // h-column owned by this lane
    float bj[4];
#pragma unroll
    for (int j = 0; j < 4; ++j) bj[j] = bvec[j * 128 + hcol];

    float creg[8], hreg[8];
#pragma unroll
    for (int r = 0; r < 8; ++r) { creg[r] = 0.f; hreg[r] = 0.f; }

    __syncthreads();

    for (int t = 0; t < TDIM; ++t) {
        // ---- stage input tile (16 x DIN) -> LDS, vectorized 16B/8B chunks -
        if (LAYER1) {
            // bf16 source: 8-element (16B) chunks; DIN/8 chunks per row
            constexpr int CPR = DIN / 8;
#pragma unroll
            for (int ci = tid; ci < BT * CPR; ci += 256) {
                int m   = ci / CPR;
                int c8  = ci - m * CPR;
                int len = sLen[m];
                int rt  = (dir == 0) ? t : ((t < len) ? (len - 1 - t) : t);
                uint4 d = *(const uint4*)(Xbf +
                    ((size_t)(b0 + m) * TDIM + rt) * (2 * HDIM) + c8 * 8);
                *(uint4*)(sX + packA_off(m, c8 * 8)) = d;
            }
        } else {
            // f32 source: 4-element chunks, convert to bf16 (8B LDS store)
            constexpr int CPR = DIN / 4;
#pragma unroll
            for (int ci = tid; ci < BT * CPR; ci += 256) {
                int m   = ci / CPR;
                int c4  = ci - m * CPR;
                int len = sLen[m];
                int rt  = (dir == 0) ? t : ((t < len) ? (len - 1 - t) : t);
                float4 v = *(const float4*)(X32 +
                    ((size_t)(b0 + m) * TDIM + rt) * DIN0 + c4 * 4);
                v4bf p;
                p[0] = (bf16_t)v.x; p[1] = (bf16_t)v.y;
                p[2] = (bf16_t)v.z; p[3] = (bf16_t)v.w;
                *(v4bf*)(sX + packA_off(m, c4 * 4)) = p;
            }
        }
        // prefetch next step's input rows into cache (global_prefetch_b8)
        if (tid < BT && t + 1 < TDIM) {
            int len = sLen[tid];
            int nt_ = (dir == 0) ? (t + 1) : ((t + 1 < len) ? (len - 2 - t) : (t + 1));
            if (LAYER1) __builtin_prefetch(Xbf + ((size_t)(b0 + tid) * TDIM + nt_) * (2 * HDIM), 0, 1);
            else        __builtin_prefetch(X32 + ((size_t)(b0 + tid) * TDIM + nt_) * DIN0, 0, 1);
        }
        __syncthreads();

        // ---- gate GEMM: acc = bias + x @ Wih^T + h @ Whh^T ----------------
        v8f acc[4];
#pragma unroll
        for (int j = 0; j < 4; ++j) {
#pragma unroll
            for (int r = 0; r < 8; ++r) acc[j][r] = bj[j];
        }
#pragma unroll
        for (int kt = 0; kt < KTI; ++kt) {
            v16bf a = *(const v16bf*)(sX + kt * 512 + L * 16);
#pragma unroll
            for (int j = 0; j < 4; ++j) {
                int   nt = w + 8 * j;
                v16bf b  = *(const v16bf*)(Wih + (size_t)(nt * KTI + kt) * 512 + L * 16);
                acc[j] = __builtin_amdgcn_wmma_f32_16x16x32_bf16(
                    false, a, false, b, (short)0, acc[j], false, false);
            }
        }
#pragma unroll
        for (int kt = 0; kt < 4; ++kt) {
            v16bf a = *(const v16bf*)(sH + kt * 512 + L * 16);
#pragma unroll
            for (int j = 0; j < 4; ++j)
                acc[j] = __builtin_amdgcn_wmma_f32_16x16x32_bf16(
                    false, a, false, whh[j][kt], (short)0, acc[j], false, false);
        }
        __syncthreads(); // all waves done reading sH before updating it

        // ---- wave-local gate nonlinearity + state update ------------------
#pragma unroll
        for (int r = 0; r < 8; ++r) {
            int   m  = r + ((L >= 16) ? 8 : 0); // C/D layout: VGPR r, lane half
            float ig = 1.f / (1.f + __expf(-acc[0][r]));
            float fg = 1.f / (1.f + __expf(-acc[1][r]));
            float gg = tanhf(acc[2][r]);
            float og = 1.f / (1.f + __expf(-acc[3][r]));
            float cn = fg * creg[r] + ig * gg;
            float hn = og * tanhf(cn);
            int   len = sLen[m];
            bool  act = t < len;
            if (act) {
                creg[r] = cn;
                hreg[r] = hn;
                sH[packA_off(m, hcol)] = (bf16_t)hn;
            }
        }

        if (!LAYER1) {
            // ---- vectorized masked hs write: 16 rows x 16 chunks of 16B ---
            __syncthreads(); // sH updates visible to the copy stage
            int   m   = tid >> 4;
            int   c8  = tid & 15;
            int   len = sLen[m];
            bool  act = t < len;
            int   rt  = (dir == 0) ? t : (act ? (len - 1 - t) : t);
            uint4 z; z.x = 0; z.y = 0; z.z = 0; z.w = 0;
            uint4 d = act ? *(const uint4*)(sH + packA_off(m, c8 * 8)) : z;
            // writing at reversed position rt IS reverse_padded(hs) for bwd
            *(uint4*)(hsOut + ((size_t)(b0 + m) * TDIM + rt) * (2 * HDIM)
                            + dir * HDIM + c8 * 8) = d;
        }
        // next iteration's sX staging is fenced by its own __syncthreads()
    }

    if (LAYER1) {
#pragma unroll
        for (int r = 0; r < 8; ++r) {
            int m = r + ((L >= 16) ? 8 : 0);
            hTOut[(size_t)dir * BDIM * HDIM + (size_t)(b0 + m) * HDIM + hcol] = hreg[r];
        }
    }
}

// ---------------------------------------------------------------------------
// Final FC + log_softmax: h = concat(hT_b, hT_f) @ Wfc^T + bfc. 3K MACs/row.
// ---------------------------------------------------------------------------
__global__ void fc_kernel(const float* __restrict__ hT,  // (2,B,H): [0]=fwd, [1]=bwd
                          const float* __restrict__ Wfc, // (12, 256)
                          const float* __restrict__ bfc, // (12,)
                          float* __restrict__ out)       // (B, 12)
{
    int b = blockIdx.x * blockDim.x + threadIdx.x;
    if (b >= BDIM) return;
    const float* hf = hT + (size_t)b * HDIM;
    const float* hb = hT + (size_t)BDIM * HDIM + (size_t)b * HDIM;
    float logits[CDIM];
    float mx = -1e30f;
    for (int c = 0; c < CDIM; ++c) {
        const float* wr = Wfc + c * (2 * HDIM);
        float s = bfc[c];
        for (int k = 0; k < HDIM; ++k) s += hb[k] * wr[k];         // h[0:128]=hT_b
        for (int k = 0; k < HDIM; ++k) s += hf[k] * wr[HDIM + k];  // h[128:256]=hT_f
        logits[c] = s;
        mx = fmaxf(mx, s);
    }
    float se = 0.f;
    for (int c = 0; c < CDIM; ++c) se += __expf(logits[c] - mx);
    float lse = mx + __logf(se);
    for (int c = 0; c < CDIM; ++c) out[b * CDIM + c] = logits[c] - lse;
}

// ---------------------------------------------------------------------------
extern "C" void kernel_launch(void* const* d_in, const int* in_sizes, int n_in,
                              void* d_out, int out_size, void* d_ws, size_t ws_size,
                              hipStream_t stream) {
    const float* X    = (const float*)d_in[0];
    const int*   lens = (const int*)d_in[1];
    const float* Wih0 = (const float*)d_in[2];
    const float* Whh0 = (const float*)d_in[3];
    const float* b0   = (const float*)d_in[4];
    const float* Wih1 = (const float*)d_in[5];
    const float* Whh1 = (const float*)d_in[6];
    const float* b1   = (const float*)d_in[7];
    const float* Wfc  = (const float*)d_in[8];
    const float* bfc  = (const float*)d_in[9];
    float* out = (float*)d_out;

    char*  ws  = (char*)d_ws;
    size_t off = 0;
    bf16_t* Wih0P = (bf16_t*)(ws + off); off += (size_t)2 * GDIM * DIN0 * sizeof(bf16_t);
    bf16_t* Whh0P = (bf16_t*)(ws + off); off += (size_t)2 * GDIM * HDIM * sizeof(bf16_t);
    bf16_t* Wih1P = (bf16_t*)(ws + off); off += (size_t)2 * GDIM * DIN1 * sizeof(bf16_t);
    bf16_t* Whh1P = (bf16_t*)(ws + off); off += (size_t)2 * GDIM * HDIM * sizeof(bf16_t);
    bf16_t* h1    = (bf16_t*)(ws + off); off += (size_t)BDIM * TDIM * 2 * HDIM * sizeof(bf16_t);
    float*  hT    = (float*)(ws + off);  off += (size_t)2 * BDIM * HDIM * sizeof(float);
    (void)ws_size; (void)in_sizes; (void)n_in; (void)out_size;

    for (int d = 0; d < 2; ++d) {
        pack_b_kernel<<<64, 256, 0, stream>>>(Wih0 + (size_t)d * GDIM * DIN0,
                                              Wih0P + (size_t)d * GDIM * DIN0, GDIM, DIN0);
        pack_b_kernel<<<64, 256, 0, stream>>>(Whh0 + (size_t)d * GDIM * HDIM,
                                              Whh0P + (size_t)d * GDIM * HDIM, GDIM, HDIM);
        pack_b_kernel<<<128, 256, 0, stream>>>(Wih1 + (size_t)d * GDIM * DIN1,
                                               Wih1P + (size_t)d * GDIM * DIN1, GDIM, DIN1);
        pack_b_kernel<<<64, 256, 0, stream>>>(Whh1 + (size_t)d * GDIM * HDIM,
                                              Whh1P + (size_t)d * GDIM * HDIM, GDIM, HDIM);
    }

    dim3 grid(BDIM / BT, 2);
    lstm_kernel<DIN0, false><<<grid, 256, 0, stream>>>(X, nullptr, Wih0P, Whh0P,
                                                       b0, lens, h1, nullptr);
    lstm_kernel<DIN1, true><<<grid, 256, 0, stream>>>(nullptr, h1, Wih1P, Whh1P,
                                                      b1, lens, nullptr, hT);
    fc_kernel<<<4, 64, 0, stream>>>(hT, Wfc, bfc, out);
}
